// KCGE_8418135900674
// MI455X (gfx1250) — matrix operand order repaired and checked by reading, not compiled, not run on hardware
//
#include <hip/hip_runtime.h>

typedef __attribute__((ext_vector_type(16))) _Float16 v16h;
typedef __attribute__((ext_vector_type(8)))  _Float16 v8h;
typedef __attribute__((ext_vector_type(8)))  float    v8f;

#define NEG_SLOPE 0.01f
#define DDIM 128

// ---------------- elementwise / graph-structure kernels ----------------

__global__ void k_zero(float* __restrict__ p, long n) {
  long i = (long)blockIdx.x * blockDim.x + threadIdx.x;
  if (i < n) p[i] = 0.0f;
}

__global__ void k_count_deg(const int* __restrict__ col, float* __restrict__ deg, int E) {
  int e = blockIdx.x * blockDim.x + threadIdx.x;
  if (e < E) atomicAdd(&deg[col[e]], 1.0f);
}

__global__ void k_deg_inv_sqrt(float* __restrict__ deg, int N) {
  int i = blockIdx.x * blockDim.x + threadIdx.x;
  if (i < N) {
    float d = deg[i];
    deg[i] = (d > 0.0f) ? rsqrtf(d) : 0.0f;
  }
}

__global__ void k_norm(const int* __restrict__ row, const int* __restrict__ col,
                       const float* __restrict__ ew, const float* __restrict__ dinv,
                       float* __restrict__ norm, int E) {
  int e = blockIdx.x * blockDim.x + threadIdx.x;
  if (e < E) norm[e] = dinv[row[e]] * dinv[col[e]] * ew[e];
}

// sum = x ; xh = f16(x)
__global__ void k_init(const float* __restrict__ x, float* __restrict__ sum,
                       _Float16* __restrict__ xh, long n) {
  long i = (long)blockIdx.x * blockDim.x + threadIdx.x;
  if (i < n) { float v = x[i]; sum[i] = v; xh[i] = (_Float16)v; }
}

// f32 -> f16 with transpose of each 128x128 relation matrix:
// WhT[t][n][k] = W[t][k][n]  (so B-fragment reads are K-contiguous)
__global__ void k_cvt_w_t(const float* __restrict__ W, _Float16* __restrict__ WhT, int n) {
  int i = blockIdx.x * blockDim.x + threadIdx.x;
  if (i < n) {
    int t   = i >> 14;            // / (128*128)
    int rem = i & 16383;
    int k   = rem >> 7;
    int col = rem & 127;
    WhT[(t << 14) + (col << 7) + k] = (_Float16)W[i];
  }
}

// z = leaky_relu(acc + b) ; sum += z ; xh = f16(z) (input to next layer's GEMM)
__global__ void k_act(const float* __restrict__ acc, const float* __restrict__ bias,
                      float* __restrict__ sum, _Float16* __restrict__ xh, long n) {
  long i = (long)blockIdx.x * blockDim.x + threadIdx.x;
  if (i < n) {
    float z = acc[i] + bias[(int)(i & (DDIM - 1))];
    z = (z > 0.0f) ? z : NEG_SLOPE * z;
    sum[i] += z;
    xh[i] = (_Float16)z;
  }
}

__global__ void k_final(const float* __restrict__ sum, float* __restrict__ out, long n) {
  long i = (long)blockIdx.x * blockDim.x + threadIdx.x;
  if (i < n) out[i] = sum[i] * 0.25f;
}

// ---------------- WMMA GEMM: xw[t] = f16(xh) @ f16(W[t]), f32 accumulate ----------------
// 4 waves / block; W[t] (32 KB f16, [n][k] layout) staged in LDS once per block.
// Each wave: one 16-row x 128-col tile. A tile (16x128) lives in 32 VGPRs for the
// whole kernel; nt-outer / kt-inner keeps a single v8f accumulator live.
// grid = (ceil(N/64), 4), block = 128.
__global__ void __launch_bounds__(128)
k_gemm_wmma(const _Float16* __restrict__ xh, const _Float16* __restrict__ WhT,
            _Float16* __restrict__ xw, int N) {
  __shared__ _Float16 Wlds[DDIM * DDIM];          // 32 KB of the 320 KB WGP LDS

  const int t = blockIdx.y;
  const _Float16* __restrict__ Wt = WhT + (size_t)t * DDIM * DDIM;

  // cooperative stage: 2048 x v8h chunks over 128 threads (16 b128 loads/thread)
  {
    const v8h* __restrict__ src = (const v8h*)Wt;
    v8h* dstl = (v8h*)Wlds;
#pragma unroll
    for (int i = 0; i < (DDIM * DDIM / 8) / 128; ++i)
      dstl[i * 128 + threadIdx.x] = src[i * 128 + threadIdx.x];
  }
  __syncthreads();

  const int wave  = threadIdx.x >> 5;
  const int maxmt = (N + 15) / 16 - 1;
  int mt = blockIdx.x * 4 + wave;
  if (mt > maxmt) mt = maxmt;                     // clamp: all waves reach the barrier
  const int m0 = mt * 16;

  const int lane = threadIdx.x & 31;
  const int nlo  = lane & 15;
  const int hsel = lane >> 4;                     // 0 = lanes 0-15, 1 = lanes 16-31

  _Float16* __restrict__ Ot = xw + (size_t)t * N * DDIM;

  int arow = m0 + nlo; if (arow >= N) arow = N - 1;
  const _Float16* __restrict__ arowp = xh + (size_t)arow * DDIM;

  // A (16-bit 16x32 per kt): lane holds row m=lane%16; elem e -> K=k0+16*hsel+e.
  // One contiguous 32B run per kt (two 16B loads).
  v16h a[4];
#pragma unroll
  for (int kt = 0; kt < 4; ++kt) {
    const _Float16* abase = arowp + kt * 32 + 16 * hsel;
    v8h alo = *(const v8h*)(abase);
    v8h ahi = *(const v8h*)(abase + 8);
    a[kt] = __builtin_shufflevector(alo, ahi, 0, 1, 2, 3, 4, 5, 6, 7,
                                    8, 9, 10, 11, 12, 13, 14, 15);
  }

  // single base pointer for the whole store epilogue; all further offsets are
  // compile-time (v*DDIM + nt*16 halves) and fold into the store immediate.
  // C/D layout: VGPR v, lane l: n = l%16, m = v + 8*(l/16).
  _Float16* __restrict__ pout = Ot + (size_t)(m0 + 8 * hsel) * DDIM + nlo;
  const bool full = (m0 + 16 <= N);               // wave-uniform (always true: N%16==0)

  const v8f vzero = {0.f, 0.f, 0.f, 0.f, 0.f, 0.f, 0.f, 0.f};
#pragma unroll
  for (int nt = 0; nt < 8; ++nt) {
    const int ncol = nt * 16 + nlo;
    v8f c = vzero;
#pragma unroll
    for (int kt = 0; kt < 4; ++kt) {
      // B (32x16): lane holds column n=lane%16; elem e -> K = kt*32 + 16*hsel + e.
      // [n][k] layout => one 32-byte contiguous LDS run (2 x ds_load_b128).
      v16h b = *(const v16h*)(Wlds + (size_t)ncol * DDIM + kt * 32 + 16 * hsel);
      c = __builtin_amdgcn_wmma_f32_16x16x32_f16(
          /*neg_a=*/false, a[kt], /*neg_b=*/false, b,
          /*c_mod=*/(short)0, c, /*reuse_a=*/false, /*reuse_b=*/false);
    }
    if (full) {
#pragma unroll
      for (int v = 0; v < 8; ++v)
        pout[v * DDIM + nt * 16] = (_Float16)c[v];
    } else {
#pragma unroll
      for (int v = 0; v < 8; ++v) {
        if (m0 + v + 8 * hsel < N)
          pout[v * DDIM + nt * 16] = (_Float16)c[v];
      }
    }
  }
}

// ---------------- edge scatter: acc[row] += norm * xw[type, col] ----------------
// 32 lanes per edge, 4 columns per lane (8B f16 gather, 4 coalesced fp32 atomics).
struct __align__(8) h4 { _Float16 a, b, c, d; };

__global__ void k_scatter(const _Float16* __restrict__ xw, const int* __restrict__ row,
                          const int* __restrict__ col, const int* __restrict__ et,
                          const float* __restrict__ norm, float* __restrict__ acc,
                          int N, int E) {
  long gid  = (long)blockIdx.x * blockDim.x + threadIdx.x;
  int  e    = (int)(gid >> 5);
  int  lane = (int)(gid & 31);
  if (e >= E) return;
  const int   r = row[e];
  const int   s = col[e];
  const int   t = et[e];
  const float w = norm[e];
  const h4 v = *(const h4*)(xw + (((size_t)t * N + s) * DDIM) + lane * 4);
  float* __restrict__ dst = acc + (size_t)r * DDIM + lane * 4;
  atomicAdd(&dst[0], (float)v.a * w);
  atomicAdd(&dst[1], (float)v.b * w);
  atomicAdd(&dst[2], (float)v.c * w);
  atomicAdd(&dst[3], (float)v.d * w);
}

// ---------------- host-side orchestration ----------------
extern "C" void kernel_launch(void* const* d_in, const int* in_sizes, int n_in,
                              void* d_out, int out_size, void* d_ws, size_t ws_size,
                              hipStream_t stream) {
  const float* x   = (const float*)d_in[0];
  const int*   ei  = (const int*)d_in[1];
  const int*   et  = (const int*)d_in[2];
  const float* ew  = (const float*)d_in[3];
  const float* Wl[3] = {(const float*)d_in[4], (const float*)d_in[6], (const float*)d_in[8]};
  const float* bl[3] = {(const float*)d_in[5], (const float*)d_in[7], (const float*)d_in[9]};

  const int  N  = in_sizes[0] / DDIM;
  const int  E  = in_sizes[2];
  const int* row = ei;
  const int* col = ei + E;
  const long nd  = (long)N * DDIM;

  // carve scratch from d_ws
  char*  ws  = (char*)d_ws;
  size_t off = 0;
  auto carve = [&](size_t bytes) -> char* {
    char* p = ws + off;
    off += (bytes + 255) & ~(size_t)255;
    return p;
  };
  float*    dinv = (float*)carve((size_t)N * 4);            // deg, then deg^-1/2 in place
  float*    nrm  = (float*)carve((size_t)E * 4);
  _Float16* xh   = (_Float16*)carve((size_t)nd * 2);
  _Float16* WhT  = (_Float16*)carve((size_t)4 * DDIM * DDIM * 2);
  _Float16* xw   = (_Float16*)carve((size_t)4 * nd * 2);
  float*    acc  = (float*)carve((size_t)nd * 4);
  float*    sum  = (float*)carve((size_t)nd * 4);
  (void)ws_size; (void)n_in; (void)out_size;

  const int TB = 256;
  // edge-structure terms (layer-invariant)
  k_zero<<<(N + TB - 1) / TB, TB, 0, stream>>>(dinv, N);
  k_count_deg<<<(E + TB - 1) / TB, TB, 0, stream>>>(col, dinv, E);
  k_deg_inv_sqrt<<<(N + TB - 1) / TB, TB, 0, stream>>>(dinv, N);
  k_norm<<<(E + TB - 1) / TB, TB, 0, stream>>>(row, col, ew, dinv, nrm, E);
  k_init<<<(unsigned)((nd + TB - 1) / TB), TB, 0, stream>>>(x, sum, xh, nd);

  for (int L = 0; L < 3; ++L) {
    k_cvt_w_t<<<(4 * DDIM * DDIM + TB - 1) / TB, TB, 0, stream>>>(Wl[L], WhT, 4 * DDIM * DDIM);
    dim3 gg((unsigned)((N + 63) / 64), 4);
    k_gemm_wmma<<<gg, 128, 0, stream>>>(xh, WhT, xw, N);
    k_zero<<<(unsigned)((nd + TB - 1) / TB), TB, 0, stream>>>(acc, nd);
    long nthr = (long)E * 32;
    k_scatter<<<(unsigned)((nthr + TB - 1) / TB), TB, 0, stream>>>(xw, row, col, et, nrm, acc, N, E);
    k_act<<<(unsigned)((nd + TB - 1) / TB), TB, 0, stream>>>(acc, bl[L], sum, xh, nd);
  }
  k_final<<<(unsigned)((nd + TB - 1) / TB), TB, 0, stream>>>(sum, (float*)d_out, nd);
}
